// AFTKVR_18133351923891
// MI455X (gfx1250) — matrix operand order repaired and checked by reading, not compiled, not run on hardware
//
#include <hip/hip_runtime.h>
#include <hip/hip_bf16.h>

// ---------------------------------------------------------------------------
// AFT-style attention (B=4, T=1024, D=256, K=63) for gfx1250 (MI455X).
// Stage 1: f32 -> bf16 conversion of x, Wq, Wk, Wv ; ew = exp(wbias)
// Stage 2: fused q/k/v GEMM via v_wmma_f32_16x16x32_bf16, epilogue writes
//          sigmoid(q), exp(k), exp(k)*v  (f32)
// Stage 3: gathered 63-term weighted reduction + final multiply (VALU, L2-bound)
// ---------------------------------------------------------------------------

typedef __bf16 v16bf __attribute__((ext_vector_type(16)));
typedef __bf16 v8bf  __attribute__((ext_vector_type(8)));
typedef float  v8f   __attribute__((ext_vector_type(8)));

#define BT   4096      // B*T rows
#define DMOD 256       // model dim
#define TT   1024      // T
#define KK   63        // keys per row

// ---- Stage 1 kernels -------------------------------------------------------

__global__ __launch_bounds__(256) void cvt_f32_bf16(const float* __restrict__ s,
                                                    __bf16* __restrict__ d, int n) {
    int i = blockIdx.x * 256 + threadIdx.x;
    if (i < n) d[i] = (__bf16)s[i];
}

__global__ __launch_bounds__(256) void exp_f32(const float* __restrict__ s,
                                               float* __restrict__ d, int n) {
    int i = blockIdx.x * 256 + threadIdx.x;
    if (i < n) d[i] = __expf(s[i]);
}

// ---- Stage 2: fused q/k/v GEMM (WMMA bf16, f32 accum) ----------------------
// Each wave: one 16x16 output tile of q, k, v (A tile shared across 3 WMMAs).
// Grid: 4096 wave-tiles / 8 waves per block = 512 blocks of 256 threads.

__global__ __launch_bounds__(256) void gemm_qkv_wmma(
    const __bf16* __restrict__ xbf,
    const __bf16* __restrict__ wq,
    const __bf16* __restrict__ wk,
    const __bf16* __restrict__ wv,
    float* __restrict__ sigq,
    float* __restrict__ ek,
    float* __restrict__ ekv) {

    const int lane = threadIdx.x & 31;
    const int wave = threadIdx.x >> 5;
    const int tile = blockIdx.x * 8 + wave;          // 0..4095
    const int m0   = (tile >> 4) << 4;               // 256 row tiles
    const int n0   = (tile & 15) << 4;               // 16 col tiles

    const int l15  = lane & 15;
    const int half = lane >> 4;                      // 0 | 1

    // A: 16x32 bf16 tile of x.  lane<16: K 0..15; lane>=16: K 16..31.
    const __bf16* aptr = xbf + (size_t)(m0 + l15) * DMOD + (half << 4);
    // B: column n of Wq^T == row (n0+l15) of W.  lane<16: K 0..15; lane>=16: K 16..31.
    const size_t boff = (size_t)(n0 + l15) * DMOD + (half << 4);
    const __bf16* qptr = wq + boff;
    const __bf16* kptr = wk + boff;
    const __bf16* vptr = wv + boff;

    v8f cq = {}; v8f ck = {}; v8f cv = {};

    #pragma unroll
    for (int kkk = 0; kkk < DMOD; kkk += 32) {
        v8bf alo = *(const v8bf*)(aptr + kkk);        // K = kb .. kb+7
        v8bf ahi = *(const v8bf*)(aptr + kkk + 8);    // K = kb+8 .. kb+15
        v16bf a  = __builtin_shufflevector(alo, ahi,
                     0,1,2,3,4,5,6,7,8,9,10,11,12,13,14,15);
        v16bf bq = *(const v16bf*)(qptr + kkk);
        v16bf bk = *(const v16bf*)(kptr + kkk);
        v16bf bv = *(const v16bf*)(vptr + kkk);
        cq = __builtin_amdgcn_wmma_f32_16x16x32_bf16(false, a, false, bq,
                                                     (short)0, cq, false, false);
        ck = __builtin_amdgcn_wmma_f32_16x16x32_bf16(false, a, false, bk,
                                                     (short)0, ck, false, false);
        cv = __builtin_amdgcn_wmma_f32_16x16x32_bf16(false, a, false, bv,
                                                     (short)0, cv, false, false);
    }

    // C/D layout: VGPR r -> M = r + 8*half, N = lane&15.
    const int ncol  = n0 + l15;
    const int rbase = m0 + (half << 3);
    #pragma unroll
    for (int r = 0; r < 8; ++r) {
        size_t off = (size_t)(rbase + r) * DMOD + ncol;
        float q = cq[r];
        float e = __expf(ck[r]);
        sigq[off] = 1.0f / (1.0f + __expf(-q));
        ek[off]   = e;
        ekv[off]  = e * cv[r];
    }
}

// ---- Stage 3: gather + weighted sum + final elementwise --------------------
// One block per (b,t); thread c owns channel c.  out = sigmoid(q) * num/den.

__global__ __launch_bounds__(256) void aft_combine(
    const float* __restrict__ sigq,
    const float* __restrict__ ek,
    const float* __restrict__ ekv,
    const float* __restrict__ ew,
    const int*   __restrict__ kidx,
    float* __restrict__ out) {

    __shared__ float sw[64];
    __shared__ int   sidx[64];

    const int b = blockIdx.x >> 10;
    const int t = blockIdx.x & (TT - 1);
    const int c = threadIdx.x;

    if (c < KK) {
        sw[c]   = ew[t * KK + c];
        sidx[c] = kidx[t * KK + c];
    }
    __syncthreads();

    const float* ekb  = ek  + (size_t)b * TT * DMOD;
    const float* ekvb = ekv + (size_t)b * TT * DMOD;

    float num = 0.0f, den = 0.0f;
    #pragma unroll 9
    for (int j = 0; j < KK; ++j) {
        const float  w = sw[j];
        const size_t o = (size_t)sidx[j] * DMOD + c;
        num = fmaf(w, ekvb[o], num);
        den = fmaf(w, ekb[o],  den);
    }

    const size_t oo = ((size_t)b * TT + t) * DMOD + c;
    out[oo] = sigq[oo] * num / den;
}

// ---------------------------------------------------------------------------

extern "C" void kernel_launch(void* const* d_in, const int* in_sizes, int n_in,
                              void* d_out, int out_size, void* d_ws, size_t ws_size,
                              hipStream_t stream) {
    const float* x     = (const float*)d_in[0];   // 4*1024*256
    const float* Wq    = (const float*)d_in[1];   // 256*256
    const float* Wk    = (const float*)d_in[2];
    const float* Wv    = (const float*)d_in[3];
    const float* wbias = (const float*)d_in[4];   // 1024*63
    const int*   kidx  = (const int*)d_in[5];     // 1024*63

    char* ws = (char*)d_ws;
    size_t off = 0;
    __bf16* xbf = (__bf16*)(ws + off); off += (size_t)BT * DMOD * sizeof(__bf16);   // 2 MB
    __bf16* wqb = (__bf16*)(ws + off); off += (size_t)DMOD * DMOD * sizeof(__bf16); // 128 KB
    __bf16* wkb = (__bf16*)(ws + off); off += (size_t)DMOD * DMOD * sizeof(__bf16);
    __bf16* wvb = (__bf16*)(ws + off); off += (size_t)DMOD * DMOD * sizeof(__bf16);
    float*  sigq = (float*)(ws + off); off += (size_t)BT * DMOD * sizeof(float);    // 4 MB
    float*  ek   = (float*)(ws + off); off += (size_t)BT * DMOD * sizeof(float);
    float*  ekv  = (float*)(ws + off); off += (size_t)BT * DMOD * sizeof(float);
    float*  ew   = (float*)(ws + off); off += (size_t)TT * KK * sizeof(float);      // 252 KB

    // Stage 1: conversions / exp(wbias)
    cvt_f32_bf16<<<(BT * DMOD + 255) / 256, 256, 0, stream>>>(x,  xbf, BT * DMOD);
    cvt_f32_bf16<<<(DMOD * DMOD + 255) / 256, 256, 0, stream>>>(Wq, wqb, DMOD * DMOD);
    cvt_f32_bf16<<<(DMOD * DMOD + 255) / 256, 256, 0, stream>>>(Wk, wkb, DMOD * DMOD);
    cvt_f32_bf16<<<(DMOD * DMOD + 255) / 256, 256, 0, stream>>>(Wv, wvb, DMOD * DMOD);
    exp_f32<<<(TT * KK + 255) / 256, 256, 0, stream>>>(wbias, ew, TT * KK);

    // Stage 2: fused q/k/v GEMM via WMMA (4096 wave tiles, 8 waves per block)
    gemm_qkv_wmma<<<512, 256, 0, stream>>>(xbf, wqb, wkb, wvb, sigq, ek, ekv);

    // Stage 3: gathered weighted reduction + epilogue
    aft_combine<<<BT, 256, 0, stream>>>(sigq, ek, ekv, ew, kidx, (float*)d_out);
}
